// GATMEncoder_38543036514339
// MI455X (gfx1250) — matrix operand order repaired and checked by reading, not compiled, not run on hardware
//
#include <hip/hip_runtime.h>
#include <hip/hip_bf16.h>

typedef __attribute__((ext_vector_type(16))) __bf16 v16bf;
typedef __attribute__((ext_vector_type(8)))  float  v8f;

#define N_NODES 8192
#define NEG_BIG (-9e15f)

// ---------- helpers ----------
__device__ __forceinline__ unsigned short f32_to_bf16(float f) {
  __bf16 b = (__bf16)f;  // native v_cvt bf16 on gfx1250
  union { __bf16 b; unsigned short u; } c; c.b = b;
  return c.u;
}
__device__ __forceinline__ float bf16_to_f32(unsigned short h) {
  union { unsigned int u; float f; } c; c.u = ((unsigned int)h) << 16;
  return c.f;
}
__device__ __forceinline__ float waveReduceMax(float v) {
#pragma unroll
  for (int o = 16; o > 0; o >>= 1) v = fmaxf(v, __shfl_xor(v, o, 32));
  return v;
}
__device__ __forceinline__ float waveReduceSum(float v) {
#pragma unroll
  for (int o = 16; o > 0; o >>= 1) v += __shfl_xor(v, o, 32);
  return v;
}

// ---------- elementwise f32 -> bf16 ----------
__global__ void cvt_f32_to_bf16(const float* __restrict__ in,
                                unsigned short* __restrict__ out, int n) {
  int i = blockIdx.x * blockDim.x + threadIdx.x;
  if (i < n) out[i] = f32_to_bf16(in[i]);
}

// ---------- max over n[] (single block) ----------
__global__ void max_reduce_kernel(const float* __restrict__ n, float* __restrict__ out) {
  __shared__ float red[8];
  float m = -3.0e38f;
  for (int i = threadIdx.x; i < N_NODES; i += 256) m = fmaxf(m, n[i]);
  m = waveReduceMax(m);
  const int wave = threadIdx.x >> 5, lane = threadIdx.x & 31;
  if (lane == 0) red[wave] = m;
  __syncthreads();
  if (threadIdx.x == 0) {
    float r = red[0];
#pragma unroll
    for (int wv = 1; wv < 8; ++wv) r = fmaxf(r, red[wv]);
    out[0] = r;
  }
}

// ---------- GEMM: hT[FOUT][N] = (A[N][K] @ W[K][FOUT])^T, bf16 WMMA ----------
// one wave (32 threads) per 16x16 output tile
template <int K, int FOUT>
__global__ void gemm_hT(const unsigned short* __restrict__ A,  // [N,K] bf16 row-major
                        const float* __restrict__ W,           // [K,FOUT] f32
                        unsigned short* __restrict__ hT) {     // [FOUT,N] bf16
  const int i0 = blockIdx.x * 16;
  const int c0 = blockIdx.y * 16;
  const int lane = threadIdx.x;
  const int half = lane >> 4;
  const int rc   = lane & 15;

  v8f acc = {};
  for (int k0 = 0; k0 < K; k0 += 32) {
    // A fragment: lane holds row (i0+rc); elems 0..7 -> k0+8*half+0..7, 8..15 -> +16
    union { v16bf v; unsigned short u[16]; } af;
    const unsigned short* arow = A + (size_t)(i0 + rc) * K + k0 + 8 * half;
#pragma unroll
    for (int e = 0; e < 8; ++e) { af.u[e] = arow[e]; af.u[8 + e] = arow[16 + e]; }
    // B fragment: lane holds col (c0+rc); elems 0..7 -> k0+8*half+0..7, 8..15 -> +16
    union { v16bf v; unsigned short u[16]; } bf;
#pragma unroll
    for (int e = 0; e < 16; ++e)
      bf.u[e] = f32_to_bf16(W[(size_t)(k0 + 8 * half + (e & 7) + 16 * (e >> 3)) * FOUT + c0 + rc]);
    acc = __builtin_amdgcn_wmma_f32_16x16x32_bf16(false, af.v, false, bf.v,
                                                  (short)0, acc, false, false);
  }
  // C layout: vgpr v, lanes0-15 -> M=v, lanes16-31 -> M=v+8; N=rc
  unsigned short* col = hT + (size_t)(c0 + rc) * N_NODES + i0 + 8 * half;
#pragma unroll
  for (int v = 0; v < 8; ++v) col[v] = f32_to_bf16(acc[v]);
}

// ---------- s = h@a_self, n = h@a_neighs from transposed bf16 h ----------
template <int FOUT>
__global__ void sn_kernel(const unsigned short* __restrict__ hT,
                          const float* __restrict__ a_self,
                          const float* __restrict__ a_neighs,
                          float* __restrict__ s, float* __restrict__ n) {
  int i = blockIdx.x * blockDim.x + threadIdx.x;
  float ss = 0.f, nn = 0.f;
#pragma unroll 4
  for (int f = 0; f < FOUT; ++f) {
    float h = bf16_to_f32(hT[(size_t)f * N_NODES + i]);
    ss += h * a_self[f];
    nn += h * a_neighs[f];
  }
  s[i] = ss; n[i] = nn;
}

// ---------- fused masked attention + (att @ h) + ELU ----------
// Softmax uses a constant per-row shift C_i = max(0, s_i + max_j n_j), which is an
// upper bound on every score (M in [0,1], leakyrelu monotone) -> no online max, no
// accumulator rescale; row sum accumulates per-lane, reduced once after the loop.
// block = 256 threads (8 waves), owns 16 output rows.
template <int FOUT>
__global__ void flash_attn(const float* __restrict__ s, const float* __restrict__ n,
                           const float* __restrict__ maxn_p,
                           const float* __restrict__ Mmat, const int* __restrict__ adj,
                           const unsigned short* __restrict__ hT,  // [FOUT][N] bf16
                           float* __restrict__ outF,               // [N][FOUT] or null
                           unsigned short* __restrict__ outB) {    // [N][FOUT] or null
  constexpr int NT = FOUT / 16;  // column tiles of 16
  __shared__ alignas(32) unsigned short Plds[32 * 16];  // A-fragment layout [lane][elem]
  __shared__ float lLds[16];

  const int i0   = blockIdx.x * 16;
  const int tid  = threadIdx.x;
  const int wave = tid >> 5;
  const int lane = tid & 31;
  const int half = lane >> 4;
  const int rc   = lane & 15;
  const int t0 = wave, t1 = wave + 8;  // this wave's column tiles (if < NT)

  const float mn = maxn_p[0];
  const float sA = s[i0 + wave];
  const float sB = s[i0 + wave + 8];
  const float shA = fmaxf(0.f, sA + mn);  // per-row softmax shift (score upper bound)
  const float shB = fmaxf(0.f, sB + mn);

  float lAacc = 0.f, lBacc = 0.f;
  v8f acc0 = {}, acc1 = {};

  // P scatter mapping (constant per lane): lane'=i+16*(k>>4), e=k&15
  const int hl = lane >> 4;
  const int ep = lane & 15;

  for (int j0 = 0; j0 < N_NODES; j0 += 32) {
    const int j = j0 + lane;
    const size_t baseA = (size_t)(i0 + wave) * N_NODES + j;
    const size_t baseB = (size_t)(i0 + wave + 8) * N_NODES + j;
    if (j0 + 32 < N_NODES) {  // CDNA5 global_prefetch_b8 of next chunk
      __builtin_prefetch(&Mmat[baseA + 32], 0, 1);
      __builtin_prefetch(&adj[baseA + 32], 0, 1);
      __builtin_prefetch(&Mmat[baseB + 32], 0, 1);
      __builtin_prefetch(&adj[baseB + 32], 0, 1);
    }
    const float nv = n[j];
    float eA = (sA + nv) * Mmat[baseA];
    float eB = (sB + nv) * Mmat[baseB];
    eA = eA > 0.f ? eA : 0.2f * eA;  // LeakyReLU(0.2)
    eB = eB > 0.f ? eB : 0.2f * eB;
    const int aA = adj[baseA], aB = adj[baseB];
    const float pA = aA ? __expf(eA - shA) : 0.f;
    const float pB = aB ? __expf(eB - shB) : 0.f;
    lAacc += pA;
    lBacc += pB;

    Plds[(wave + 16 * hl) * 16 + ep]     = f32_to_bf16(pA);
    Plds[(wave + 8 + 16 * hl) * 16 + ep] = f32_to_bf16(pB);
    __syncthreads();

    // acc += P(16x32) @ H(32x16) per owned tile
    const v16bf pv = *(const v16bf*)&Plds[lane * 16];
    if (t0 < NT) {
      const v16bf bv = *(const v16bf*)&hT[(size_t)(t0 * 16 + rc) * N_NODES + j0 + 16 * half];
      acc0 = __builtin_amdgcn_wmma_f32_16x16x32_bf16(false, pv, false, bv,
                                                     (short)0, acc0, false, false);
    }
    if (t1 < NT) {
      const v16bf bv = *(const v16bf*)&hT[(size_t)(t1 * 16 + rc) * N_NODES + j0 + 16 * half];
      acc1 = __builtin_amdgcn_wmma_f32_16x16x32_bf16(false, pv, false, bv,
                                                     (short)0, acc1, false, false);
    }
    __syncthreads();
  }

  const float lA = waveReduceSum(lAacc);
  const float lB = waveReduceSum(lBacc);
  if (lane == 0) { lLds[wave] = lA; lLds[wave + 8] = lB; }
  __syncthreads();

  // epilogue: normalize, ELU, store row-major
#pragma unroll
  for (int t = 0; t < 2; ++t) {
    const int tile = (t == 0) ? t0 : t1;
    if (tile >= NT) continue;
    const v8f acc = (t == 0) ? acc0 : acc1;
    const int c = tile * 16 + rc;
#pragma unroll
    for (int v = 0; v < 8; ++v) {
      const int row = v + 8 * half;
      float x = acc[v] / lLds[row];
      x = x > 0.f ? x : (__expf(x) - 1.f);  // ELU
      const size_t idx = (size_t)(i0 + row) * FOUT + c;
      if (outF) outF[idx] = x;
      if (outB) outB[idx] = f32_to_bf16(x);
    }
  }
}

// ---------- launch ----------
extern "C" void kernel_launch(void* const* d_in, const int* in_sizes, int n_in,
                              void* d_out, int out_size, void* d_ws, size_t ws_size,
                              hipStream_t stream) {
  const float* x   = (const float*)d_in[0];
  const int*   adj = (const int*)d_in[1];
  const float* Mm  = (const float*)d_in[2];
  const float* W0  = (const float*)d_in[3];
  const float* as0 = (const float*)d_in[4];
  const float* an0 = (const float*)d_in[5];
  const float* W1  = (const float*)d_in[6];
  const float* as1 = (const float*)d_in[7];
  const float* an1 = (const float*)d_in[8];
  float* out = (float*)d_out;

  char* w = (char*)d_ws;
  unsigned short* xb    = (unsigned short*)w; w += (size_t)N_NODES * 512 * 2;
  unsigned short* hT0   = (unsigned short*)w; w += (size_t)256 * N_NODES * 2;
  float* s0             = (float*)w;          w += (size_t)N_NODES * 4;
  float* n0             = (float*)w;          w += (size_t)N_NODES * 4;
  unsigned short* act0b = (unsigned short*)w; w += (size_t)N_NODES * 256 * 2;
  unsigned short* hT1   = (unsigned short*)w; w += (size_t)64 * N_NODES * 2;
  float* s1             = (float*)w;          w += (size_t)N_NODES * 4;
  float* n1             = (float*)w;          w += (size_t)N_NODES * 4;
  float* maxn0          = (float*)w;          w += 256;
  float* maxn1          = (float*)w;          w += 256;

  // ---- layer 1 ----
  cvt_f32_to_bf16<<<(N_NODES * 512) / 256, 256, 0, stream>>>(x, xb, N_NODES * 512);
  gemm_hT<512, 256><<<dim3(N_NODES / 16, 256 / 16), 32, 0, stream>>>(xb, W0, hT0);
  sn_kernel<256><<<N_NODES / 256, 256, 0, stream>>>(hT0, as0, an0, s0, n0);
  max_reduce_kernel<<<1, 256, 0, stream>>>(n0, maxn0);
  flash_attn<256><<<N_NODES / 16, 256, 0, stream>>>(s0, n0, maxn0, Mm, adj, hT0,
                                                    nullptr, act0b);
  // ---- layer 2 ----
  gemm_hT<256, 64><<<dim3(N_NODES / 16, 64 / 16), 32, 0, stream>>>(act0b, W1, hT1);
  sn_kernel<64><<<N_NODES / 256, 256, 0, stream>>>(hT1, as1, an1, s1, n1);
  max_reduce_kernel<<<1, 256, 0, stream>>>(n1, maxn1);
  flash_attn<64><<<N_NODES / 16, 256, 0, stream>>>(s1, n1, maxn1, Mm, adj, hT1,
                                                   out, nullptr);
}
